// MiniMaxText01MoE_8478265442852
// MI455X (gfx1250) — matrix operand order repaired and checked by reading, not compiled, not run on hardware
//
#include <hip/hip_runtime.h>
#include <hip/hip_bf16.h>

// ---------------- problem constants ----------------
constexpr int E_EXP = 8;
constexpr int TOPK  = 2;
constexpr int H_DIM = 2048;
constexpr int I_DIM = 5632;
constexpr int T_TOK = 4096;

typedef __attribute__((ext_vector_type(16))) __bf16 bf16x16;
typedef __attribute__((ext_vector_type(8)))  float  floatx8;

// LDS row stride (elements) for 32-wide K tiles; padded to dodge bank conflicts
// while keeping 16B alignment of fragment chunks ((40*2)=80B row pitch, chunk
// offsets 0/16/32/48 bytes are 16B aligned).
constexpr int LDSK = 40;

// ---------------- WMMA fragment helpers (per CDNA5 ISA layouts) ----
// A (16x32 bf16): lane m<16 -> row m, K 0..7 (v0-3) & 16..23 (v4-7);
//                 lane m+16 -> row m, K 8..15 & 24..31.  Pairs are contiguous.
__device__ inline bf16x16 load_frag_a(const __bf16* S, int row0, int lane) {
  const int m  = lane & 15;
  const int kb = (lane >> 4) * 8;
  const __bf16* p = S + (row0 + m) * LDSK + kb;
  bf16x16 r;
#pragma unroll
  for (int i = 0; i < 8; ++i) r[i] = p[i];
#pragma unroll
  for (int i = 0; i < 8; ++i) r[8 + i] = p[16 + i];
  return r;
}

// B (32x16 bf16): lane n<16 -> col n, K 0..7 & 16..23; lane n+16 -> K 8..15 & 24..31.
// LDS holds B as [n][k] (weights are N-major in memory: w[n][k]).
__device__ inline bf16x16 load_frag_b(const __bf16* S, int col0, int lane) {
  const int n  = lane & 15;
  const int kb = (lane >> 4) * 8;
  const __bf16* p = S + (col0 + n) * LDSK + kb;
  bf16x16 r;
#pragma unroll
  for (int i = 0; i < 8; ++i) r[i] = p[i];
#pragma unroll
  for (int i = 0; i < 8; ++i) r[8 + i] = p[16 + i];
  return r;
}

__device__ inline floatx8 wmma_bf16(bf16x16 a, bf16x16 b, floatx8 c) {
  return __builtin_amdgcn_wmma_f32_16x16x32_bf16(
      /*neg_a=*/false, a, /*neg_b=*/false, b,
      /*c_mod=*/(short)0, c, /*reuse_a=*/false, /*reuse_b=*/false);
}

// ---------------- kernel 0: zero output ----------------
__global__ __launch_bounds__(256) void moe_zero(float* __restrict__ p, int n) {
  int i = blockIdx.x * 256 + threadIdx.x;
  if (i < n) p[i] = 0.0f;
}

// ---------------- kernel 1: gating (one wave per token) ----------------
__global__ __launch_bounds__(256) void moe_gate(const float* __restrict__ hs,
                                                const float* __restrict__ gw,
                                                float* __restrict__ dense_w) {
  const int lane = threadIdx.x & 31;
  const int wid  = threadIdx.x >> 5;
  const int t    = blockIdx.x * 8 + wid;
  if (t >= T_TOK) return;
  const float* hrow = hs + (size_t)t * H_DIM;

  float logit[E_EXP];
#pragma unroll
  for (int e = 0; e < E_EXP; ++e) {
    const float* grow = gw + (size_t)e * H_DIM;
    float s = 0.0f;
    for (int k = lane; k < H_DIM; k += 32) s += hrow[k] * grow[k];
    // butterfly reduce across wave32
#pragma unroll
    for (int off = 16; off > 0; off >>= 1) s += __shfl_xor(s, off, 32);
    logit[e] = s;
  }
  if (lane == 0) {
    float m = logit[0];
#pragma unroll
    for (int e = 1; e < E_EXP; ++e) m = fmaxf(m, logit[e]);
    float p[E_EXP]; float sum = 0.0f;
#pragma unroll
    for (int e = 0; e < E_EXP; ++e) { p[e] = __expf(logit[e] - m); sum += p[e]; }
    const float inv = 1.0f / sum;
#pragma unroll
    for (int e = 0; e < E_EXP; ++e) p[e] *= inv;
    // top-2, ties -> lower index (matches lax.top_k)
    int i0 = 0; float v0 = p[0];
#pragma unroll
    for (int e = 1; e < E_EXP; ++e) if (p[e] > v0) { v0 = p[e]; i0 = e; }
    int i1 = -1; float v1 = -1.0f;
#pragma unroll
    for (int e = 0; e < E_EXP; ++e) if (e != i0 && p[e] > v1) { v1 = p[e]; i1 = e; }
    const float norm = 1.0f / (v0 + v1);
    float* dw = dense_w + (size_t)t * E_EXP;
#pragma unroll
    for (int e = 0; e < E_EXP; ++e) dw[e] = 0.0f;
    dw[i0] = v0 * norm;
    dw[i1] = v1 * norm;
  }
}

// ---------------- kernel 2: deterministic per-expert token compaction --------
__global__ __launch_bounds__(256) void moe_build_lists(const float* __restrict__ dense_w,
                                                       int* __restrict__ tokens,
                                                       float* __restrict__ wlist,
                                                       int* __restrict__ counts) {
  const int e   = blockIdx.x;
  const int tid = threadIdx.x;
  __shared__ int scan[256];
  __shared__ int base;
  if (tid == 0) base = 0;
  __syncthreads();
  for (int start = 0; start < T_TOK; start += 256) {
    const int t = start + tid;
    const float w = dense_w[(size_t)t * E_EXP + e];
    const int flag = (w > 0.0f) ? 1 : 0;
    scan[tid] = flag;
    __syncthreads();
    // Hillis-Steele inclusive scan
    for (int off = 1; off < 256; off <<= 1) {
      int v = (tid >= off) ? scan[tid - off] : 0;
      __syncthreads();
      scan[tid] += v;
      __syncthreads();
    }
    if (flag) {
      const int pos = base + scan[tid] - 1;
      tokens[e * T_TOK + pos] = t;
      wlist[e * T_TOK + pos]  = w;
    }
    __syncthreads();
    if (tid == 0) base += scan[255];
    __syncthreads();
  }
  if (tid == 0) counts[e] = base;
}

// ---------------- kernel 3: cast hidden states to bf16 ----------------
__global__ __launch_bounds__(256) void moe_cast_bf16(const float* __restrict__ x,
                                                     __bf16* __restrict__ y, int n) {
  int i = blockIdx.x * 256 + threadIdx.x;
  if (i < n) y[i] = (__bf16)x[i];
}

// ---------------- kernel 4: per-expert GEMM1 + SiLU fuse ----------------
// h[t, n] = silu(hs@w13g.T) * (hs@w13u.T); tokens gathered per expert.
// Block tile: M=128 tokens x N=64 h-cols; 8 waves in 4(M) x 2(N); per wave
// 32x32 -> 2x2 WMMA tiles for g and for u.  K-loop is software-pipelined:
// next step's global loads are issued before the current step's WMMAs.
__global__ __launch_bounds__(256) void moe_gemm1(int e,
                                                 const int* __restrict__ counts,
                                                 const int* __restrict__ tokens,
                                                 const __bf16* __restrict__ hs,
                                                 const float* __restrict__ w13,
                                                 __bf16* __restrict__ h_buf) {
  const int count = counts[e];
  const int M0 = blockIdx.x * 128;
  if (M0 >= count) return;          // block-uniform -> EXEC stays all-ones
  const int N0 = blockIdx.y * 64;
  const float* w13e = w13 + (size_t)e * (size_t)(2 * I_DIM) * H_DIM;
  const int* toks = tokens + e * T_TOK;

  __shared__ __align__(16) __bf16 As[128 * LDSK];
  __shared__ __align__(16) __bf16 Bgs[64 * LDSK];
  __shared__ __align__(16) __bf16 Bus[64 * LDSK];

  const int tid  = threadIdx.x;
  const int lane = tid & 31;
  const int wid  = tid >> 5;
  const int wm   = wid >> 1;  // 0..3 -> M offset wm*32
  const int wn   = wid & 1;   // 0..1 -> N offset wn*32

  // A staging: thread -> (row, 16-elem half)
  const int arow  = tid >> 1;
  const int ahalf = tid & 1;
  const int aslot = (M0 + arow < count) ? (M0 + arow) : (count - 1);
  const int atok  = toks[aslot];
  // B staging: thread -> (n-row, 8-elem quarter)
  const int brow = tid >> 2;
  const int bq   = tid & 3;

  const __bf16* aGlob = hs + (size_t)atok * H_DIM + ahalf * 16;
  const float*  gGlob = w13e + (size_t)(N0 + brow) * H_DIM + bq * 8;
  const float*  uGlob = w13e + ((size_t)I_DIM + N0 + brow) * H_DIM + bq * 8;

  floatx8 accg[2][2], accu[2][2];
#pragma unroll
  for (int i = 0; i < 2; ++i)
#pragma unroll
    for (int j = 0; j < 2; ++j) {
      accg[i][j] = (floatx8)(0.0f);
      accu[i][j] = (floatx8)(0.0f);
    }

  // register-staged tiles (software pipeline)
  uint4 aR0, aR1;
  float gR[8], uR[8];
  {
    const uint4* s = reinterpret_cast<const uint4*>(aGlob);
    aR0 = s[0]; aR1 = s[1];
#pragma unroll
    for (int i = 0; i < 8; ++i) { gR[i] = gGlob[i]; uR[i] = uGlob[i]; }
  }

  for (int k0 = 0; k0 < H_DIM; k0 += 32) {
    __syncthreads();  // previous step's LDS reads done
    // drain registers -> LDS (A bf16 pass-through, B f32 -> bf16)
    {
      uint4* dst = reinterpret_cast<uint4*>(&As[arow * LDSK + ahalf * 16]);
      dst[0] = aR0; dst[1] = aR1;
      __bf16* dg = &Bgs[brow * LDSK + bq * 8];
      __bf16* du = &Bus[brow * LDSK + bq * 8];
#pragma unroll
      for (int i = 0; i < 8; ++i) dg[i] = (__bf16)gR[i];
#pragma unroll
      for (int i = 0; i < 8; ++i) du[i] = (__bf16)uR[i];
    }
    __syncthreads();  // LDS tiles ready

    // issue next step's global loads before the WMMA block (latency overlap)
    if (k0 + 32 < H_DIM) {
      const int kn = k0 + 32;
      const uint4* s = reinterpret_cast<const uint4*>(aGlob + kn);
      aR0 = s[0]; aR1 = s[1];
#pragma unroll
      for (int i = 0; i < 8; ++i) { gR[i] = gGlob[kn + i]; uR[i] = uGlob[kn + i]; }
    }

    bf16x16 a[2], bg[2], bu[2];
#pragma unroll
    for (int mt = 0; mt < 2; ++mt) a[mt] = load_frag_a(As, wm * 32 + mt * 16, lane);
#pragma unroll
    for (int nt = 0; nt < 2; ++nt) {
      bg[nt] = load_frag_b(Bgs, wn * 32 + nt * 16, lane);
      bu[nt] = load_frag_b(Bus, wn * 32 + nt * 16, lane);
    }
#pragma unroll
    for (int mt = 0; mt < 2; ++mt)
#pragma unroll
      for (int nt = 0; nt < 2; ++nt) {
        accg[mt][nt] = wmma_bf16(a[mt], bg[nt], accg[mt][nt]);
        accu[mt][nt] = wmma_bf16(a[mt], bu[nt], accu[mt][nt]);
      }
  }

  // Epilogue: h = silu(g)*u -> bf16  (fast v_rcp instead of precise division)
  const int m_hi = (lane >> 4) << 3;  // C layout: lanes 16..31 hold rows +8
  const int nn   = lane & 15;
#pragma unroll
  for (int mt = 0; mt < 2; ++mt)
#pragma unroll
    for (int nt = 0; nt < 2; ++nt)
#pragma unroll
      for (int v = 0; v < 8; ++v) {
        const float g = accg[mt][nt][v];
        const float u = accu[mt][nt][v];
        const float sig = __builtin_amdgcn_rcpf(1.0f + __expf(-g));
        const float hv = g * sig * u;
        const int row = M0 + wm * 32 + mt * 16 + m_hi + v;
        const int col = N0 + wn * 32 + nt * 16 + nn;
        h_buf[(size_t)row * I_DIM + col] = (__bf16)hv;
      }
}

// ---------------- kernel 5: per-expert GEMM2, weight, accumulate -------------
// out[token, j] += route_w * (h @ w2.T)[slot, j]
// Block tile: M=128 x N=128; 8 waves in 4(M) x 2(N); wave = 32x64 -> 2x4 tiles.
// Same software pipeline as GEMM1.
__global__ __launch_bounds__(256) void moe_gemm2(int e,
                                                 const int* __restrict__ counts,
                                                 const int* __restrict__ tokens,
                                                 const float* __restrict__ wlist,
                                                 const __bf16* __restrict__ h_buf,
                                                 const float* __restrict__ w2,
                                                 float* __restrict__ out) {
  const int count = counts[e];
  const int M0 = blockIdx.x * 128;
  if (M0 >= count) return;
  const int N0 = blockIdx.y * 128;
  const float* w2e = w2 + (size_t)e * (size_t)H_DIM * I_DIM;
  const int* toks = tokens + e * T_TOK;
  const float* wl = wlist + e * T_TOK;

  __shared__ __align__(16) __bf16 As[128 * LDSK];
  __shared__ __align__(16) __bf16 Bs[128 * LDSK];

  const int tid  = threadIdx.x;
  const int lane = tid & 31;
  const int wid  = tid >> 5;
  const int wm   = wid >> 1;  // 0..3
  const int wn   = wid & 1;   // 0..1

  const int arow  = tid >> 1;
  const int ahalf = tid & 1;

  const __bf16* aGlob = h_buf + (size_t)(M0 + arow) * I_DIM + ahalf * 16;
  const float*  bGlob = w2e + (size_t)(N0 + arow) * I_DIM + ahalf * 16;

  floatx8 acc[2][4];
#pragma unroll
  for (int i = 0; i < 2; ++i)
#pragma unroll
    for (int j = 0; j < 4; ++j) acc[i][j] = (floatx8)(0.0f);

  uint4 aR0, aR1;
  float bR[16];
  {
    const uint4* s = reinterpret_cast<const uint4*>(aGlob);
    aR0 = s[0]; aR1 = s[1];
#pragma unroll
    for (int i = 0; i < 16; ++i) bR[i] = bGlob[i];
  }

  for (int k0 = 0; k0 < I_DIM; k0 += 32) {
    __syncthreads();
    {
      uint4* dst = reinterpret_cast<uint4*>(&As[arow * LDSK + ahalf * 16]);
      dst[0] = aR0; dst[1] = aR1;
      __bf16* db = &Bs[arow * LDSK + ahalf * 16];
#pragma unroll
      for (int i = 0; i < 16; ++i) db[i] = (__bf16)bR[i];
    }
    __syncthreads();

    if (k0 + 32 < I_DIM) {
      const int kn = k0 + 32;
      const uint4* s = reinterpret_cast<const uint4*>(aGlob + kn);
      aR0 = s[0]; aR1 = s[1];
#pragma unroll
      for (int i = 0; i < 16; ++i) bR[i] = bGlob[kn + i];
    }

    bf16x16 a[2], b[4];
#pragma unroll
    for (int mt = 0; mt < 2; ++mt) a[mt] = load_frag_a(As, wm * 32 + mt * 16, lane);
#pragma unroll
    for (int nt = 0; nt < 4; ++nt) b[nt] = load_frag_b(Bs, wn * 64 + nt * 16, lane);
#pragma unroll
    for (int mt = 0; mt < 2; ++mt)
#pragma unroll
      for (int nt = 0; nt < 4; ++nt) acc[mt][nt] = wmma_bf16(a[mt], b[nt], acc[mt][nt]);
  }

  const int m_hi = (lane >> 4) << 3;
  const int nn   = lane & 15;
#pragma unroll
  for (int mt = 0; mt < 2; ++mt)
#pragma unroll
    for (int v = 0; v < 8; ++v) {
      const int row = M0 + wm * 32 + mt * 16 + m_hi + v;
      if (row < count) {
        const int tok  = toks[row];
        const float rw = wl[row];
        float* orow = out + (size_t)tok * H_DIM;
#pragma unroll
        for (int nt = 0; nt < 4; ++nt) {
          const int col = N0 + wn * 64 + nt * 16 + nn;
          orow[col] += rw * acc[mt][nt][v];  // experts run sequentially
        }
      }
    }
}

// ---------------- host launcher ----------------
extern "C" void kernel_launch(void* const* d_in, const int* in_sizes, int n_in,
                              void* d_out, int out_size, void* d_ws, size_t ws_size,
                              hipStream_t stream) {
  const float* hs   = (const float*)d_in[0];  // (T, H)
  const float* gw   = (const float*)d_in[1];  // (E, H)
  const float* w13  = (const float*)d_in[2];  // (E, 2I, H)
  const float* w2   = (const float*)d_in[3];  // (E, H, I)
  float* out = (float*)d_out;                 // (T, H)

  auto align256 = [](size_t x) { return (x + 255) & ~(size_t)255; };
  char* ws = (char*)d_ws;
  size_t off = 0;
  float* dense_w = (float*)(ws + off); off = align256(off + (size_t)T_TOK * E_EXP * 4);
  int*   counts  = (int*)(ws + off);   off = align256(off + (size_t)E_EXP * 4);
  int*   tokens  = (int*)(ws + off);   off = align256(off + (size_t)E_EXP * T_TOK * 4);
  float* wlist   = (float*)(ws + off); off = align256(off + (size_t)E_EXP * T_TOK * 4);
  __bf16* hs_bf  = (__bf16*)(ws + off); off = align256(off + (size_t)T_TOK * H_DIM * 2);
  __bf16* h_buf  = (__bf16*)(ws + off); off = align256(off + (size_t)T_TOK * I_DIM * 2);
  (void)ws_size; (void)in_sizes; (void)n_in; (void)out_size;

  const int nOut = T_TOK * H_DIM;
  moe_zero<<<(nOut + 255) / 256, 256, 0, stream>>>(out, nOut);
  moe_gate<<<T_TOK / 8, 256, 0, stream>>>(hs, gw, dense_w);
  moe_build_lists<<<E_EXP, 256, 0, stream>>>(dense_w, tokens, wlist, counts);
  moe_cast_bf16<<<(nOut + 255) / 256, 256, 0, stream>>>(hs, hs_bf, nOut);

  dim3 g1(T_TOK / 128, I_DIM / 64);    // 32 x 88 (blocks past count early-exit)
  dim3 g2(T_TOK / 128, H_DIM / 128);   // 32 x 16
  for (int e = 0; e < E_EXP; ++e) {
    moe_gemm1<<<g1, 256, 0, stream>>>(e, counts, tokens, hs_bf, w13, h_buf);
    moe_gemm2<<<g2, 256, 0, stream>>>(e, counts, tokens, wlist, h_buf, w2, out);
  }
}
